// RelPartialLearnableDecoderLayer_51754355917018
// MI455X (gfx1250) — matrix operand order repaired and checked by reading, not compiled, not run on hardware
//
#include <hip/hip_runtime.h>
#include <hip/hip_bf16.h>

// ---------------------------------------------------------------------------
// Transformer-XL decoder layer (GTrXL gating) for MI455X / gfx1250.
// All matmuls run through a BF16 WMMA (v_wmma_f32_16x16x32_bf16) GEMM with a
// double-buffered LDS pipeline (stage tile k+1 while computing tile k).
// qlen=512 mlen=512 klen=1024 bsz=4 d_model=1024 d_inner=4096 NH=16 DH=64
// ---------------------------------------------------------------------------

typedef __bf16 bf16_t;
typedef __attribute__((ext_vector_type(16))) __bf16 v16bf;
typedef __attribute__((ext_vector_type(8)))  __bf16 v8bf;
typedef __attribute__((ext_vector_type(8)))  float  v8f;

#define GF_OUT_BF16 1
#define GF_RELU     2
#define GF_ACCUM    4

// ---------------------------------------------------------------------------
// Generic LDS-staged BF16 WMMA GEMM:  C[M,N] = A[M,K] @ B[K,N]  (+Cin)(+bias)
// A row-major bf16, B row-major bf16 (transposed into LDS), C f32 or bf16.
// Batched via blockIdx.z with per-operand strides; B may be shared across a
// sub-batch via modB (ptrB += (bz % modB) * strideB).
// ---------------------------------------------------------------------------
template<int BM, int BN, int WM, int WN>
__global__ __launch_bounds__((BM / WM) * (BN / WN) * 32)
void gemm_bf16_kernel(const bf16_t* __restrict__ A, const bf16_t* __restrict__ B,
                      void* __restrict__ Cout, const float* __restrict__ Cin,
                      const float* __restrict__ bias,
                      int K, int lda, int ldb, int ldc,
                      long strideA, long strideB, long strideC,
                      int modB, int flags)
{
    constexpr int NWAVES = (BM / WM) * (BN / WN);
    constexpr int NT     = NWAVES * 32;
    constexpr int PITCH  = 32 + 8;          // bf16 elements; keeps 16B alignment
    constexpr int MF = WM / 16, NF = WN / 16;
    constexpr int AIT = (BM * 32) / (NT * 8);   // staging chunks (16B) per thread
    constexpr int BIT = (32 * BN) / (NT * 8);
    static_assert(WM % 16 == 0 && WN % 16 == 0, "frag");
    static_assert(AIT >= 1 && BIT >= 1, "staging");

    __shared__ bf16_t As[2][BM * PITCH];    // [BM][32] (+pad), double-buffered
    __shared__ bf16_t Bs[2][BN * PITCH];    // transposed: [BN][32] (+pad)

    const int tid   = threadIdx.x;
    const int lane  = tid & 31;
    const int lrow  = lane & 15;
    const int lhalf = lane >> 4;
    const int wave  = tid >> 5;
    constexpr int WAVES_N = BN / WN;
    const int wm0 = (wave / WAVES_N) * WM;
    const int wn0 = (wave % WAVES_N) * WN;

    const int  bz   = blockIdx.z;
    const long aOff = (long)bz * strideA;
    const long bOff = (long)(bz % modB) * strideB;
    const long cOff = (long)bz * strideC;
    const int  m0   = blockIdx.y * BM;
    const int  n0   = blockIdx.x * BN;

    // --- precomputed staging descriptors (advance by constant stride per tile)
    const bf16_t* aPtr[AIT];
    int           aDst[AIT];
    #pragma unroll
    for (int it = 0; it < AIT; ++it) {
        int idx = (tid + it * NT) * 8;
        int row = idx >> 5, col = idx & 31;
        aPtr[it] = A + aOff + (long)(m0 + row) * lda + col;
        aDst[it] = row * PITCH + col;
    }
    const bf16_t* bPtr[BIT];
    int           bDst[BIT];
    #pragma unroll
    for (int it = 0; it < BIT; ++it) {
        int idx = (tid + it * NT) * 8;
        int row = idx / BN, col = idx % BN;
        bPtr[it] = B + bOff + (long)row * ldb + n0 + col;
        bDst[it] = col * PITCH + row;           // transposed [n][k]
    }
    const long ldb32 = (long)ldb * 32;

    // --- stage one BK=32 tile into buffer `buf`; advances pointers.
    // Chunks are held in ext-vector registers (no address taken -> no scratch).
    auto stage = [&](int buf) {
        v8bf av[AIT], bv[BIT];
        #pragma unroll
        for (int it = 0; it < AIT; ++it) { av[it] = *(const v8bf*)aPtr[it]; aPtr[it] += 32; }
        #pragma unroll
        for (int it = 0; it < BIT; ++it) { bv[it] = *(const v8bf*)bPtr[it]; bPtr[it] += ldb32; }
        #pragma unroll
        for (int it = 0; it < AIT; ++it) {
            *(v8bf*)&As[buf][aDst[it]] = av[it];
            __builtin_prefetch((const void*)aPtr[it], 0, 3);   // tile after next
        }
        #pragma unroll
        for (int it = 0; it < BIT; ++it) {
            #pragma unroll
            for (int e = 0; e < 8; ++e) Bs[buf][bDst[it] + e * PITCH] = bv[it][e];
            __builtin_prefetch((const void*)bPtr[it], 0, 3);
        }
    };

    v8f acc[MF][NF];
    v8f vzero = {};
    #pragma unroll
    for (int im = 0; im < MF; ++im)
        #pragma unroll
        for (int in = 0; in < NF; ++in) acc[im][in] = vzero;

    const int KT = K >> 5;
    stage(0);
    for (int kt = 0; kt < KT; ++kt) {
        const int buf = kt & 1;
        __syncthreads();                    // publish `buf`, retire reads of buf^1
        if (kt + 1 < KT) stage(buf ^ 1);

        // --- fragments (ISA 7.12.2 16-bit 16x32 A layout; B mirrored)
        v16bf af[MF], bfr[NF];
        #pragma unroll
        for (int im = 0; im < MF; ++im) {
            const bf16_t* ap = &As[buf][(wm0 + im * 16 + lrow) * PITCH];
            #pragma unroll
            for (int j = 0; j < 8; ++j) {
                int k = (j < 4) ? (lhalf * 8 + 2 * j) : (16 + lhalf * 8 + 2 * (j - 4));
                af[im][2 * j]     = ap[k];
                af[im][2 * j + 1] = ap[k + 1];
            }
        }
        #pragma unroll
        for (int in = 0; in < NF; ++in) {
            const bf16_t* bp = &Bs[buf][(wn0 + in * 16 + lrow) * PITCH];
            #pragma unroll
            for (int j = 0; j < 8; ++j) {
                int k = (j < 4) ? (lhalf * 8 + 2 * j) : (16 + lhalf * 8 + 2 * (j - 4));
                bfr[in][2 * j]     = bp[k];
                bfr[in][2 * j + 1] = bp[k + 1];
            }
        }
        #pragma unroll
        for (int im = 0; im < MF; ++im)
            #pragma unroll
            for (int in = 0; in < NF; ++in)
                acc[im][in] = __builtin_amdgcn_wmma_f32_16x16x32_bf16(
                    false, af[im], false, bfr[in], (short)0, acc[im][in], false, false);
    }

    // --- epilogue: D layout VGPR v -> M = v + 8*lhalf, N = lrow
    float*  Cf = (float*)Cout;
    bf16_t* Cb = (bf16_t*)Cout;
    #pragma unroll
    for (int im = 0; im < MF; ++im) {
        #pragma unroll
        for (int in = 0; in < NF; ++in) {
            int   col = n0 + wn0 + in * 16 + lrow;
            float bv  = bias ? bias[col] : 0.f;
            #pragma unroll
            for (int v = 0; v < 8; ++v) {
                int  rowm = m0 + wm0 + im * 16 + v + 8 * lhalf;
                long cidx = cOff + (long)rowm * ldc + col;
                float x = acc[im][in][v] + bv;
                if (flags & GF_ACCUM) x += Cin[cidx];
                if (flags & GF_RELU)  x = fmaxf(x, 0.f);
                if (flags & GF_OUT_BF16) Cb[cidx] = (bf16_t)x;
                else                     Cf[cidx] = x;
            }
        }
    }
}

// ---------------------------------------------------------------------------
// Elementwise / reduction helper kernels
// ---------------------------------------------------------------------------
__global__ void cvt_bf16_kernel(const float* __restrict__ s, bf16_t* __restrict__ d, long n) {
    long i = (long)blockIdx.x * blockDim.x + threadIdx.x;
    if (i < n) d[i] = (bf16_t)s[i];
}

__global__ void relu_bf16_kernel(const float* __restrict__ s, bf16_t* __restrict__ d, long n) {
    long i = (long)blockIdx.x * blockDim.x + threadIdx.x;
    if (i < n) d[i] = (bf16_t)fmaxf(s[i], 0.f);
}

// rows selected from srcA (row < split) or srcB (row - split); LN -> bf16
__global__ void layernorm_bf16_kernel(const float* __restrict__ srcA,
                                      const float* __restrict__ srcB, int split,
                                      const float* __restrict__ gamma,
                                      const float* __restrict__ beta,
                                      bf16_t* __restrict__ dst, int cols)
{
    int row = blockIdx.x, tid = threadIdx.x;
    const float* src = (row < split) ? (srcA + (long)row * cols)
                                     : (srcB + (long)(row - split) * cols);
    float s = 0.f, ss = 0.f;
    for (int c = tid; c < cols; c += blockDim.x) { float v = src[c]; s += v; ss += v * v; }
    __shared__ float sh1[256], sh2[256];
    sh1[tid] = s; sh2[tid] = ss; __syncthreads();
    for (int o = 128; o > 0; o >>= 1) {
        if (tid < o) { sh1[tid] += sh1[tid + o]; sh2[tid] += sh2[tid + o]; }
        __syncthreads();
    }
    float mu   = sh1[0] / cols;
    float var  = sh2[0] / cols - mu * mu;
    float rstd = rsqrtf(var + 1e-5f);
    bf16_t* out = dst + (long)row * cols;
    for (int c = tid; c < cols; c += blockDim.x)
        out[c] = (bf16_t)((src[c] - mu) * rstd * gamma[c] + beta[c]);
}

// qw/qr = bf16(q + bias); heads bf16 [4096,3072], q = rows 512.. , cols 0..1023
__global__ void prep_q_kernel(const bf16_t* __restrict__ heads,
                              const float* __restrict__ rwb, const float* __restrict__ rrb,
                              bf16_t* __restrict__ qw, bf16_t* __restrict__ qr)
{
    long idx = (long)blockIdx.x * blockDim.x + threadIdx.x;   // [b][n][i][d]
    if (idx >= 4L * 16 * 512 * 64) return;
    int d = idx & 63, i = (idx >> 6) & 511, n = (idx >> 15) & 15, b = (int)(idx >> 19);
    float qv = (float)heads[((long)((512 + i) * 4 + b)) * 3072 + n * 64 + d];
    qw[idx] = (bf16_t)(qv + rwb[n * 64 + d]);
    qr[idx] = (bf16_t)(qv + rrb[n * 64 + d]);
}

// kT[b][n][d][j] = heads[j*4+b][1024 + n*64 + d]
__global__ void prep_kT_kernel(const bf16_t* __restrict__ heads, bf16_t* __restrict__ kT) {
    long idx = (long)blockIdx.x * blockDim.x + threadIdx.x;
    if (idx >= 4L * 16 * 64 * 1024) return;
    int j = idx & 1023, d = (idx >> 10) & 63, n = (idx >> 16) & 15, b = (int)(idx >> 20);
    kT[idx] = heads[((long)(j * 4 + b)) * 3072 + 1024 + n * 64 + d];
}

// v[b][n][j][d] = heads[j*4+b][2048 + n*64 + d]
__global__ void prep_v_kernel(const bf16_t* __restrict__ heads, bf16_t* __restrict__ vv) {
    long idx = (long)blockIdx.x * blockDim.x + threadIdx.x;
    if (idx >= 4L * 16 * 1024 * 64) return;
    int d = idx & 63, j = (idx >> 6) & 1023, n = (idx >> 16) & 15, b = (int)(idx >> 20);
    vv[idx] = heads[((long)(j * 4 + b)) * 3072 + 2048 + n * 64 + d];
}

// rkT[n][d][j] = rk[j][n*64+d]
__global__ void prep_rkT_kernel(const bf16_t* __restrict__ rk, bf16_t* __restrict__ rkT) {
    long idx = (long)blockIdx.x * blockDim.x + threadIdx.x;
    if (idx >= 16L * 64 * 1024) return;
    int j = idx & 1023, d = (idx >> 10) & 63, n = (int)(idx >> 16);
    rkT[idx] = rk[(long)j * 1024 + n * 64 + d];
}

// score = (AC[i,j] + BDraw[i, j+511-i]) * 1/8, masked for j > i+512.
// softmax over j; writes bf16 prob and per-row entropy partial.
__global__ void softmax_entropy_kernel(const bf16_t* __restrict__ AC,
                                       const bf16_t* __restrict__ BD,
                                       bf16_t* __restrict__ prob,
                                       float* __restrict__ ent_part)
{
    const int  i     = blockIdx.x;       // 0..511
    const int  batch = blockIdx.y;       // b*16+n
    const long base  = ((long)batch * 512 + i) * 1024;
    const int  tid   = threadIdx.x;
    const int  jmax  = i + 512;
    const float NEG  = -__builtin_inff();

    float s[4], mx = NEG;
    #pragma unroll
    for (int t = 0; t < 4; ++t) {
        int j = tid + t * 256;
        if (j <= jmax) {
            int jj = j + 511 - i;        // rel_shift gather (valid region only)
            s[t] = ((float)AC[base + j] + (float)BD[base + jj]) * 0.125f;
            mx = fmaxf(mx, s[t]);
        } else s[t] = NEG;
    }
    __shared__ float sh[256];
    sh[tid] = mx; __syncthreads();
    for (int o = 128; o > 0; o >>= 1) { if (tid < o) sh[tid] = fmaxf(sh[tid], sh[tid + o]); __syncthreads(); }
    mx = sh[0]; __syncthreads();

    float e[4], sum = 0.f;
    #pragma unroll
    for (int t = 0; t < 4; ++t) { e[t] = (s[t] == NEG) ? 0.f : __expf(s[t] - mx); sum += e[t]; }
    sh[tid] = sum; __syncthreads();
    for (int o = 128; o > 0; o >>= 1) { if (tid < o) sh[tid] += sh[tid + o]; __syncthreads(); }
    float Z = sh[0]; __syncthreads();
    float invZ = 1.f / Z, logZ = __logf(Z);

    float ent = 0.f;
    #pragma unroll
    for (int t = 0; t < 4; ++t) {
        int j = tid + t * 256;
        float pv = e[t] * invZ;
        prob[base + j] = (bf16_t)pv;                       // 0 in masked region
        if (s[t] != NEG) ent += pv * (mx + logZ - s[t]);   // -p*log(p), 0-safe
    }
    sh[tid] = ent; __syncthreads();
    for (int o = 128; o > 0; o >>= 1) { if (tid < o) sh[tid] += sh[tid + o]; __syncthreads(); }
    if (tid == 0) ent_part[(long)batch * 512 + i] = sh[0];
}

__global__ void entropy_reduce_kernel(const float* __restrict__ ent_part, float* __restrict__ out) {
    int n = blockIdx.x, tid = threadIdx.x;
    float s = 0.f;
    for (int b = 0; b < 4; ++b)
        for (int i = tid; i < 512; i += 256) s += ent_part[((long)(b * 16 + n)) * 512 + i];
    __shared__ float sh[256];
    sh[tid] = s; __syncthreads();
    for (int o = 128; o > 0; o >>= 1) { if (tid < o) sh[tid] += sh[tid + o]; __syncthreads(); }
    if (tid == 0) out[n] = sh[0] * (1.f / 2048.f);
}

// attn_flat[(i*4+b)][n*64+d] = ctx[b][n][i][d]
__global__ void ctx_rearrange_kernel(const float* __restrict__ ctx, bf16_t* __restrict__ flat) {
    long idx = (long)blockIdx.x * blockDim.x + threadIdx.x;   // dst index
    if (idx >= 2048L * 1024) return;
    int d = idx & 63, n = (idx >> 6) & 15, b = (idx >> 10) & 3, i = (int)(idx >> 12);
    flat[idx] = (bf16_t)ctx[(((long)(b * 16 + n)) * 512 + i) * 64 + d];
}

// rx = bf16(sigmoid(rpre) * x)
__global__ void rx_mul_kernel(const float* __restrict__ rpre, const float* __restrict__ x,
                              bf16_t* __restrict__ rx, long n) {
    long i = (long)blockIdx.x * blockDim.x + threadIdx.x;
    if (i < n) rx[i] = (bf16_t)((1.f / (1.f + __expf(-rpre[i]))) * x[i]);
}

// out = (1-z)*x + z*tanh(hpre), z = sigmoid(zpre)
__global__ void gate_combine_kernel(const float* __restrict__ zpre, const float* __restrict__ hpre,
                                    const float* __restrict__ x, float* __restrict__ out, long n) {
    long i = (long)blockIdx.x * blockDim.x + threadIdx.x;
    if (i < n) {
        float z = 1.f / (1.f + __expf(-zpre[i]));
        out[i] = (1.f - z) * x[i] + z * tanhf(hpre[i]);
    }
}

// ---------------------------------------------------------------------------
// Host orchestration
// ---------------------------------------------------------------------------
static inline void launch_gemm128(hipStream_t st, const bf16_t* A, const bf16_t* B, void* C,
                                  const float* Cin, const float* bias,
                                  int M, int N, int K, int lda, int ldb, int ldc,
                                  long sA, long sB, long sC, int modB, int batches, int flags)
{
    dim3 g(N / 128, M / 128, batches), b(256);
    gemm_bf16_kernel<128, 128, 64, 32><<<g, b, 0, st>>>(A, B, C, Cin, bias, K, lda, ldb, ldc,
                                                        sA, sB, sC, modB, flags);
}
static inline void launch_gemm64(hipStream_t st, const bf16_t* A, const bf16_t* B, void* C,
                                 const float* Cin, const float* bias,
                                 int M, int N, int K, int lda, int ldb, int ldc,
                                 long sA, long sB, long sC, int modB, int batches, int flags)
{
    dim3 g(N / 64, M / 64, batches), b(128);
    gemm_bf16_kernel<64, 64, 32, 32><<<g, b, 0, st>>>(A, B, C, Cin, bias, K, lda, ldb, ldc,
                                                      sA, sB, sC, modB, flags);
}

extern "C" void kernel_launch(void* const* d_in, const int* in_sizes, int n_in,
                              void* d_out, int out_size, void* d_ws, size_t ws_size,
                              hipStream_t stream)
{
    (void)in_sizes; (void)n_in; (void)out_size; (void)ws_size;
    const float* w      = (const float*)d_in[0];
    const float* mems   = (const float*)d_in[1];
    const float* r      = (const float*)d_in[2];
    const float* rwb    = (const float*)d_in[3];
    const float* rrb    = (const float*)d_in[4];
    // d_in[5] attn_mask: deterministic (j > i+512), recomputed on device
    const float* ln_a_w = (const float*)d_in[6];
    const float* ln_a_b = (const float*)d_in[7];
    const float* W_qkv  = (const float*)d_in[8];
    const float* W_r    = (const float*)d_in[9];
    const float* W_o    = (const float*)d_in[10];
    const float* ln2_w  = (const float*)d_in[11];
    const float* ln2_b  = (const float*)d_in[12];
    const float* ff_W1  = (const float*)d_in[13];
    const float* ff_b1  = (const float*)d_in[14];
    const float* ff_W2  = (const float*)d_in[15];
    const float* ff_b2  = (const float*)d_in[16];
    const float* gW[12];
    for (int i = 0; i < 6; ++i) gW[i]     = (const float*)d_in[17 + i]; // gm: Wr Ur Uz Wg Ug Wz
    const float* gm_bz = (const float*)d_in[23];
    for (int i = 0; i < 6; ++i) gW[6 + i] = (const float*)d_in[24 + i]; // gp: Wr Ur Uz Wg Ug Wz
    const float* gp_bz = (const float*)d_in[30];

    // ---- workspace carve -------------------------------------------------
    char* p = (char*)d_ws;
    auto alloc = [&](size_t bytes) -> void* {
        void* ret = (void*)p; p += (bytes + 255) & ~(size_t)255; return ret;
    };
    bf16_t* b_wqkv = (bf16_t*)alloc(3145728L * 2);
    bf16_t* b_wr   = (bf16_t*)alloc(1048576L * 2);
    bf16_t* b_wo   = (bf16_t*)alloc(1048576L * 2);
    bf16_t* b_ff1  = (bf16_t*)alloc(4194304L * 2);
    bf16_t* b_ff2  = (bf16_t*)alloc(4194304L * 2);
    bf16_t* b_g[12];
    for (int i = 0; i < 12; ++i) b_g[i] = (bf16_t*)alloc(1048576L * 2);
    bf16_t* b_rbf  = (bf16_t*)alloc(1048576L * 2);
    bf16_t* b_wbf  = (bf16_t*)alloc(2097152L * 2);
    bf16_t* b_xln  = (bf16_t*)alloc(4194304L * 2);                  // LN(cat) [4096,1024]
    bf16_t* b_heads= (bf16_t*)alloc(12582912L * 2);                 // [4096,3072]
    bf16_t* b_rk   = (bf16_t*)alloc(1048576L * 2);                  // [1024,1024]
    bf16_t* b_qw   = (bf16_t*)alloc(2097152L * 2);                  // [b,n,i,d]
    bf16_t* b_qr   = (bf16_t*)alloc(2097152L * 2);
    bf16_t* b_kT   = (bf16_t*)alloc(4194304L * 2);                  // [b,n,d,j]
    bf16_t* b_v    = (bf16_t*)alloc(4194304L * 2);                  // [b,n,j,d]
    bf16_t* b_rkT  = (bf16_t*)alloc(1048576L * 2);                  // [n,d,j]
    bf16_t* b_AC   = (bf16_t*)alloc(33554432L * 2);                 // [b,n,i,j]
    bf16_t* b_BD   = (bf16_t*)alloc(33554432L * 2);
    bf16_t* b_prob = (bf16_t*)alloc(33554432L * 2);
    float*  b_ent  = (float*)alloc(32768L * 4);                     // [b*n, i]
    float*  b_ctx  = (float*)alloc(2097152L * 4);                   // [b,n,i,d]
    bf16_t* b_aflat= (bf16_t*)alloc(2097152L * 2);                  // [(i,b), nd]
    float*  b_aout = (float*)alloc(2097152L * 4);                   // attn_out f32
    bf16_t* b_y    = (bf16_t*)alloc(2097152L * 2);                  // relu(y) bf16
    float*  b_G1   = (float*)alloc(2097152L * 4);
    float*  b_G2   = (float*)alloc(2097152L * 4);
    float*  b_G3   = (float*)alloc(2097152L * 4);
    bf16_t* b_rx   = (bf16_t*)alloc(2097152L * 2);
    float*  b_o1   = (float*)alloc(2097152L * 4);
    bf16_t* b_o1b  = (bf16_t*)alloc(2097152L * 2);
    bf16_t* b_x2   = (bf16_t*)alloc(2097152L * 2);                  // LN2 bf16
    bf16_t* b_f1   = (bf16_t*)alloc(8388608L * 2);                  // [2048,4096]
    float*  b_ffo  = (float*)alloc(2097152L * 4);

    const int BIG = 1 << 30;
    auto cvt = [&](const float* s, bf16_t* d, long n) {
        cvt_bf16_kernel<<<dim3((unsigned)((n + 255) / 256)), dim3(256), 0, stream>>>(s, d, n);
    };

    // ---- weight / input conversions to bf16 ------------------------------
    cvt(W_qkv, b_wqkv, 3145728);
    cvt(W_r,   b_wr,   1048576);
    cvt(W_o,   b_wo,   1048576);
    cvt(ff_W1, b_ff1,  4194304);
    cvt(ff_W2, b_ff2,  4194304);
    for (int i = 0; i < 12; ++i) cvt(gW[i], b_g[i], 1048576);
    cvt(r, b_rbf, 1048576);
    cvt(w, b_wbf, 2097152);

    // ---- LN(cat) and projections -----------------------------------------
    layernorm_bf16_kernel<<<dim3(4096), dim3(256), 0, stream>>>(
        mems, w, 2048, ln_a_w, ln_a_b, b_xln, 1024);
    // heads = LN(cat) @ W_qkv   [4096,3072]
    launch_gemm128(stream, b_xln, b_wqkv, b_heads, nullptr, nullptr,
                   4096, 3072, 1024, 1024, 3072, 3072, 0, 0, 0, BIG, 1, GF_OUT_BF16);
    // r_k = r @ W_r             [1024,1024]
    launch_gemm128(stream, b_rbf, b_wr, b_rk, nullptr, nullptr,
                   1024, 1024, 1024, 1024, 1024, 1024, 0, 0, 0, BIG, 1, GF_OUT_BF16);

    // ---- attention prep ---------------------------------------------------
    prep_q_kernel  <<<dim3(8192),  dim3(256), 0, stream>>>(b_heads, rwb, rrb, b_qw, b_qr);
    prep_kT_kernel <<<dim3(16384), dim3(256), 0, stream>>>(b_heads, b_kT);
    prep_v_kernel  <<<dim3(16384), dim3(256), 0, stream>>>(b_heads, b_v);
    prep_rkT_kernel<<<dim3(4096),  dim3(256), 0, stream>>>(b_rk, b_rkT);

    // ---- batched score GEMMs (64 batches of 512x1024x64) -----------------
    launch_gemm128(stream, b_qw, b_kT, b_AC, nullptr, nullptr,
                   512, 1024, 64, 64, 1024, 1024,
                   512L * 64, 64L * 1024, 512L * 1024, BIG, 64, GF_OUT_BF16);
    launch_gemm128(stream, b_qr, b_rkT, b_BD, nullptr, nullptr,
                   512, 1024, 64, 64, 1024, 1024,
                   512L * 64, 64L * 1024, 512L * 1024, 16 /* B indexed by head */, 64, GF_OUT_BF16);

    // ---- rel-shift + mask + softmax + entropy ----------------------------
    softmax_entropy_kernel<<<dim3(512, 64), dim3(256), 0, stream>>>(b_AC, b_BD, b_prob, b_ent);

    // ---- context = prob @ v  (64 batches of 512x64x1024) -----------------
    launch_gemm64(stream, b_prob, b_v, b_ctx, nullptr, nullptr,
                  512, 64, 1024, 1024, 64, 64,
                  512L * 1024, 1024L * 64, 512L * 64, BIG, 64, 0);
    ctx_rearrange_kernel<<<dim3(8192), dim3(256), 0, stream>>>(b_ctx, b_aflat);

    // ---- output projection -----------------------------------------------
    launch_gemm128(stream, b_aflat, b_wo, b_aout, nullptr, nullptr,
                   2048, 1024, 1024, 1024, 1024, 1024, 0, 0, 0, BIG, 1, 0);
    relu_bf16_kernel<<<dim3(8192), dim3(256), 0, stream>>>(b_aout, b_y, 2097152);

    // ---- GRU gate gm: x = w, y = relu(attn_out) --------------------------
    // z_pre = y@Wz + bz + x@Uz ; r_pre = y@Wr + x@Ur ; h_pre = y@Wg + (r*x)@Ug
    launch_gemm128(stream, b_y,   b_g[5], b_G1, nullptr, gm_bz,
                   2048, 1024, 1024, 1024, 1024, 1024, 0, 0, 0, BIG, 1, 0);
    launch_gemm128(stream, b_wbf, b_g[2], b_G1, b_G1, nullptr,
                   2048, 1024, 1024, 1024, 1024, 1024, 0, 0, 0, BIG, 1, GF_ACCUM);
    launch_gemm128(stream, b_y,   b_g[0], b_G2, nullptr, nullptr,
                   2048, 1024, 1024, 1024, 1024, 1024, 0, 0, 0, BIG, 1, 0);
    launch_gemm128(stream, b_wbf, b_g[1], b_G2, b_G2, nullptr,
                   2048, 1024, 1024, 1024, 1024, 1024, 0, 0, 0, BIG, 1, GF_ACCUM);
    rx_mul_kernel<<<dim3(8192), dim3(256), 0, stream>>>(b_G2, w, b_rx, 2097152);
    launch_gemm128(stream, b_y,  b_g[3], b_G3, nullptr, nullptr,
                   2048, 1024, 1024, 1024, 1024, 1024, 0, 0, 0, BIG, 1, 0);
    launch_gemm128(stream, b_rx, b_g[4], b_G3, b_G3, nullptr,
                   2048, 1024, 1024, 1024, 1024, 1024, 0, 0, 0, BIG, 1, GF_ACCUM);
    gate_combine_kernel<<<dim3(8192), dim3(256), 0, stream>>>(b_G1, b_G3, w, b_o1, 2097152);
    cvt(b_o1, b_o1b, 2097152);

    // ---- FFN --------------------------------------------------------------
    layernorm_bf16_kernel<<<dim3(2048), dim3(256), 0, stream>>>(
        b_o1, b_o1, 0, ln2_w, ln2_b, b_x2, 1024);
    launch_gemm128(stream, b_x2, b_ff1, b_f1, nullptr, ff_b1,
                   2048, 4096, 1024, 1024, 4096, 4096, 0, 0, 0, BIG, 1, GF_RELU | GF_OUT_BF16);
    launch_gemm128(stream, b_f1, b_ff2, b_ffo, nullptr, ff_b2,
                   2048, 1024, 4096, 4096, 1024, 1024, 0, 0, 0, BIG, 1, 0);
    relu_bf16_kernel<<<dim3(8192), dim3(256), 0, stream>>>(b_ffo, b_y, 2097152);

    // ---- GRU gate gp: x = o1, y = relu(ff) -> writes d_out ---------------
    float* out = (float*)d_out;
    launch_gemm128(stream, b_y,   b_g[11], b_G1, nullptr, gp_bz,
                   2048, 1024, 1024, 1024, 1024, 1024, 0, 0, 0, BIG, 1, 0);
    launch_gemm128(stream, b_o1b, b_g[8],  b_G1, b_G1, nullptr,
                   2048, 1024, 1024, 1024, 1024, 1024, 0, 0, 0, BIG, 1, GF_ACCUM);
    launch_gemm128(stream, b_y,   b_g[6],  b_G2, nullptr, nullptr,
                   2048, 1024, 1024, 1024, 1024, 1024, 0, 0, 0, BIG, 1, 0);
    launch_gemm128(stream, b_o1b, b_g[7],  b_G2, b_G2, nullptr,
                   2048, 1024, 1024, 1024, 1024, 1024, 0, 0, 0, BIG, 1, GF_ACCUM);
    rx_mul_kernel<<<dim3(8192), dim3(256), 0, stream>>>(b_G2, b_o1, b_rx, 2097152);
    launch_gemm128(stream, b_y,  b_g[9],  b_G3, nullptr, nullptr,
                   2048, 1024, 1024, 1024, 1024, 1024, 0, 0, 0, BIG, 1, 0);
    launch_gemm128(stream, b_rx, b_g[10], b_G3, b_G3, nullptr,
                   2048, 1024, 1024, 1024, 1024, 1024, 0, 0, 0, BIG, 1, GF_ACCUM);
    gate_combine_kernel<<<dim3(8192), dim3(256), 0, stream>>>(b_G1, b_G3, b_o1, out, 2097152);

    // ---- per-head attention entropy -> tail of d_out ---------------------
    entropy_reduce_kernel<<<dim3(16), dim3(256), 0, stream>>>(b_ent, out + 2097152);
}
